// AudioDeviceModel_90074054131717
// MI455X (gfx1250) — matrix operand order repaired and checked by reading, not compiled, not run on hardware
//
#include <hip/hip_runtime.h>

typedef __attribute__((ext_vector_type(16))) _Float16 v16h;
typedef __attribute__((ext_vector_type(8)))  _Float16 v8h;
typedef __attribute__((ext_vector_type(8)))  float    v8f;

#define T_TOTAL  4096
#define WIN      2174   // 128 + 2*(1+2+...+512) : receptive-field tail
#define WINP     (WIN + 16)   // +16 zero/pad rows for ragged WMMA tiles
#define FRAME    128
#define NTHREADS 1024
#define NWAVES   (NTHREADS/32)
#define NLAYERS  10

__global__ __launch_bounds__(NTHREADS)
void wavenet_tail_wmma(const float* __restrict__ x,
                       const float* __restrict__ c0_kernel,   // (3,1,8)
                       const float* __restrict__ c_kernels,   // (9,3,8,8)
                       const float* __restrict__ c_biases,    // (10,8)
                       const float* __restrict__ io_kernels,  // (9,1,8,8)
                       const float* __restrict__ io_biases,   // (9,8)
                       const float* __restrict__ mixer_kernel,// (1,80,1)
                       const float* __restrict__ mixer_bias,  // (1,)
                       float* __restrict__ out, int T)
{
  __shared__ float xs[WIN];                        // input tail window (f32)
  __shared__ float h32[WIN*8];                     // residual trunk, full precision
  __shared__ __align__(16) _Float16 h16[WINP*8];   // f16 shadow (WMMA A); rows >= WIN are zero
  __shared__ __align__(16) _Float16 ybuf[WINP*8];  // activations; rows >= WIN are scratch pad
  __shared__ float mixacc[FRAME];
  __shared__ float wconv[192];
  __shared__ float cbias[8];
  __shared__ float wio[64];
  __shared__ float bio[8];
  __shared__ float wmix[8];

  const int tid = threadIdx.x;
  const int b   = blockIdx.x;
  const float* xb = x + (size_t)b * T + (T - WIN);

  // warm L2 with the (shared, reused-by-all-blocks) weight tables
  if (tid < 64) __builtin_prefetch(c_kernels  + tid * 32, 0, 1);   // 1728 f32
  if (tid >= 64 && tid < 96) __builtin_prefetch(io_kernels + (tid - 64) * 32, 0, 1);

  for (int t = tid; t < WIN; t += NTHREADS) xs[t] = xb[t];
  // zero the 16 pad rows of h16 (rows WIN..WIN+15): ragged-tile reads hit true zeros
  if (tid < 16 * 8) h16[WIN * 8 + tid] = (_Float16)0.f;
  if (tid < FRAME) mixacc[tid] = 0.f;
  __syncthreads();

  // y-tail length needed per layer (receptive-field bookkeeping, computed offline)
  const int M_tab[NLAYERS] = {2172,2168,2160,2144,2112,2048,1920,1664,1152,128};

  const int lane = tid & 31;
  const int wv   = tid >> 5;
  const int half = lane >> 4;   // which 16-lane half of the wave
  const int idx  = lane & 15;   // A: row M ; B/C/D: column N

  for (int layer = 0; layer < NLAYERS; ++layer) {
    // ---- stage per-layer constants into LDS (disjoint tid ranges) ----
    if (layer == 0) { if (tid < 24) wconv[tid] = c0_kernel[tid]; }
    else if (tid < 192) wconv[tid] = c_kernels[(layer-1)*192 + tid];
    if (tid >= 192 && tid < 200) cbias[tid-192] = c_biases[layer*8 + (tid-192)];
    if (tid >= 200 && tid < 208) wmix [tid-200] = mixer_kernel[layer*8 + (tid-200)];
    if (layer < 9) {
      if (tid >= 256 && tid < 320) wio[tid-256] = io_kernels[layer*64 + (tid-256)];
      if (tid >= 320 && tid < 328) bio[tid-320] = io_biases [layer*8  + (tid-320)];
    }
    __syncthreads();

    const int M    = M_tab[layer];
    const int base = WIN - M;
    const int d    = 1 << layer;

    if (layer == 0) {
      // single input channel: K=3 -> plain FMAs
      for (int t = base + tid; t < WIN; t += NTHREADS) {
        const float x0 = xs[t-2], x1 = xs[t-1], x2 = xs[t];
        #pragma unroll
        for (int o = 0; o < 8; ++o) {
          float v = cbias[o] + x0*wconv[o] + x1*wconv[8+o] + x2*wconv[16+o];
          ybuf[t*8+o] = (_Float16)(v > 0.f ? v : 0.f);
        }
      }
    } else {
      // ---- B operand: weights K[j][cin][o], K-dim index = j*8+cin (24 used of 32) ----
      // B layout: lane column N=idx, element e -> K = (e&7) + 8*half + 16*(e>>3)
      v16h bmat;
      #pragma unroll
      for (int e = 0; e < 16; ++e) {
        const int K = (e&7) + 8*half + 16*(e>>3);
        const int j = K >> 3;                 // tap index; j==3 is zero padding
        const float w = (idx < 8 && j < 3) ? wconv[j*64 + (e&7)*8 + idx] : 0.f;
        bmat[e] = (_Float16)w;
      }
      const float cb = (idx < 8) ? cbias[idx] : 0.f;

      const int ntiles = (M + 15) >> 4;
      for (int tile = wv; tile < ntiles; tile += NWAVES) {
        const int t0 = base + tile*16;
        const int tA = t0 + idx;                  // may run into zeroed pad rows
        const int tL = tA - (half ? d : 2*d);     // always >= 0 by construction
        const int tH = half ? WIN : tA;           // half=1 -> zero row (K=24..31 pad)
        // A layout (16-bit 16x32): lanes 0-15: e0-7 -> tap0 (t-2d), e8-15 -> tap2 (t)
        //                          lanes16-31: e0-7 -> tap1 (t-d),  e8-15 -> zeros
        const v8h loRow = *(const v8h*)(h16 + tL*8);
        const v8h hiRow = *(const v8h*)(h16 + tH*8);
        v16h amat;
        #pragma unroll
        for (int e = 0; e < 8; ++e) {
          amat[e]   = loRow[e];
          amat[8+e] = hiRow[e];
        }
        v8f acc = {};
        acc = __builtin_amdgcn_wmma_f32_16x16x32_f16(
                  /*neg_a=*/false, amat, /*neg_b=*/false, bmat,
                  /*c_mod=*/(short)0, acc, /*reuse_a=*/false, /*reuse_b=*/false);
        // D layout: VGPR v -> row M = v + 8*half, column N = idx.
        // No row guard: ragged rows land in ybuf pad rows (never read back).
        if (idx < 8) {
          #pragma unroll
          for (int v = 0; v < 8; ++v) {
            const int tr = t0 + v + 8*half;
            const float val = acc[v] + cb;
            ybuf[tr*8 + idx] = (_Float16)(val > 0.f ? val : 0.f);
          }
        }
      }
    }
    __syncthreads();

    // ---- mixer accumulation over the last FRAME steps ----
    if (tid < FRAME) {
      const int t = WIN - FRAME + tid;
      float s = mixacc[tid];
      const v8h yr = *(const v8h*)(ybuf + t*8);
      #pragma unroll
      for (int c = 0; c < 8; ++c) s += (float)yr[c] * wmix[c];
      mixacc[tid] = s;
    }

    // ---- 1x1 io projection + residual (f32 trunk, f16 shadow) ----
    if (layer < 9) {
      for (int t = base + tid; t < WIN; t += NTHREADS) {
        const v8h yr = *(const v8h*)(ybuf + t*8);
        float yf[8];
        #pragma unroll
        for (int o = 0; o < 8; ++o) yf[o] = (float)yr[o];
        #pragma unroll
        for (int c = 0; c < 8; ++c) {
          float s = bio[c] + ((layer == 0) ? xs[t] : h32[t*8+c]);
          #pragma unroll
          for (int o = 0; o < 8; ++o) s += yf[o]*wio[o*8+c];
          h32[t*8+c] = s;
          h16[t*8+c] = (_Float16)s;
        }
      }
    }
    __syncthreads();
  }

  if (tid < FRAME)
    out[(size_t)b*FRAME + tid] = mixacc[tid] + mixer_bias[0];
}

extern "C" void kernel_launch(void* const* d_in, const int* in_sizes, int n_in,
                              void* d_out, int out_size, void* d_ws, size_t ws_size,
                              hipStream_t stream)
{
  const float* x   = (const float*)d_in[0];
  const float* c0  = (const float*)d_in[1];
  const float* ck  = (const float*)d_in[2];
  const float* cb  = (const float*)d_in[3];
  const float* iok = (const float*)d_in[4];
  const float* iob = (const float*)d_in[5];
  const float* mk  = (const float*)d_in[6];
  const float* mb  = (const float*)d_in[7];
  float* out = (float*)d_out;
  const int B = in_sizes[0] / T_TOTAL;   // 64
  (void)n_in; (void)out_size; (void)d_ws; (void)ws_size;
  wavenet_tail_wmma<<<dim3(B), dim3(NTHREADS), 0, stream>>>(
      x, c0, ck, cb, iok, iob, mk, mb, out, T_TOTAL);
}
